// MultiHeadSpikingAttention_69724499083936
// MI455X (gfx1250) — matrix operand order repaired and checked by reading, not compiled, not run on hardware
//
#include <hip/hip_runtime.h>
#include <hip/hip_bf16.h>
#include <math.h>

typedef __attribute__((ext_vector_type(16))) _Float16 v16h;
typedef __attribute__((ext_vector_type(8)))  _Float16 v8h;
typedef __attribute__((ext_vector_type(4)))  _Float16 v4h;
typedef __attribute__((ext_vector_type(2)))  _Float16 v2h;
typedef __attribute__((ext_vector_type(8)))  float    v8f;

union V16U { v16h v; v8h h[2]; };

// Problem constants (match reference)
constexpr int Bn = 2, Sn = 2048, En = 1024, Hn = 16, Dn = 64;
constexpr int WINc = 128, STRIDEc = 64, NGLOBc = 4;
constexpr int GMAX = 64;   // >= 35 possible strided/global columns per head

// ---------------- fragment loaders (vectorized LDS access) ----------------
// A-matrix 16x32: lane<16 holds K={0..15}, lane>=16 K={16..31} (matches frag_B).
__device__ __forceinline__ v16h frag_A(const _Float16* rowbase, int lane) {
  const int lh = (lane >> 4) << 4;
  V16U u;
  u.h[0] = *(const v8h*)(rowbase + lh);
  u.h[1] = *(const v8h*)(rowbase + 8 + lh);
  return u.v;
}
// B-matrix 32x16: lane group holds 16 contiguous K values -> 2x ds_load_b128.
__device__ __forceinline__ v16h frag_B(const _Float16* p) {
  V16U u;
  u.h[0] = *(const v8h*)(p);
  u.h[1] = *(const v8h*)(p + 8);
  return u.v;
}
// packed f32->f16 conversion (v_cvt_pk_rtz_f16_f32): 2 converts per float4
__device__ __forceinline__ v4h pack4(float4 f) {
  v2h lo = __builtin_bit_cast(v2h, __builtin_amdgcn_cvt_pkrtz(f.x, f.y));
  v2h hi = __builtin_bit_cast(v2h, __builtin_amdgcn_cvt_pkrtz(f.z, f.w));
  return __builtin_shufflevector(lo, hi, 0, 1, 2, 3);
}

// =====================================================================================
// GEMM: C[M,N] = A[M,K] @ B[K,N] + bias[N]  (f32 in/out, f16 WMMA compute)
// Block 128 threads (4 waves), block tile 128x64, wave tile 32x64 (2 A-frags share
// 4 B-frags -> 1.5 LDS loads per WMMA). Register-staged software pipeline: next
// K-tile's global loads are issued before compute, stored to LDS after compute.
// =====================================================================================
__global__ __launch_bounds__(128) void gemm_bias_f16(
    const float* __restrict__ A, const float* __restrict__ Bm,
    const float* __restrict__ bias, float* __restrict__ C,
    int M, int N, int K)
{
  constexpr int ASTR = 40;   // 32 k + 8 pad (halves)
  constexpr int BSTR = 40;   // transposed layout [n][k]
  __shared__ __align__(32) _Float16 As [128 * ASTR];
  __shared__ __align__(32) _Float16 BsT[ 64 * BSTR];

  const int t = threadIdx.x;
  const int L = t & 31;
  const int w = t >> 5;
  const int lh16 = (L >> 4) << 4;
  const int m0 = blockIdx.y * 128;
  const int n0 = blockIdx.x * 64;

  v8f acc[2][4];
  for (int m2 = 0; m2 < 2; m2++)
    for (int c = 0; c < 4; c++)
      for (int r = 0; r < 8; r++) acc[m2][c][r] = 0.f;

  float4 ar[8], br[4];
  auto gload = [&](int k0) {               // global -> registers (no waits at issue)
    for (int i = 0; i < 8; i++) {          // A tile 128x32 = 1024 float4
      int slot = t + i * 128;
      int rr = slot >> 3, c4 = slot & 7;
      ar[i] = *(const float4*)&A[(size_t)(m0 + rr) * K + k0 + c4 * 4];
    }
    for (int i = 0; i < 4; i++) {          // B tile 32x64 = 512 float4
      int slot = t + i * 128;
      int kk = slot >> 4, n4 = slot & 15;
      br[i] = *(const float4*)&Bm[(size_t)(k0 + kk) * N + n0 + n4 * 4];
    }
  };
  auto sstore = [&]() {                    // registers -> LDS (f16)
    for (int i = 0; i < 8; i++) {
      int slot = t + i * 128;
      int rr = slot >> 3, c4 = slot & 7;
      *(v4h*)&As[rr * ASTR + c4 * 4] = pack4(ar[i]);
    }
    for (int i = 0; i < 4; i++) {
      int slot = t + i * 128;
      int kk = slot >> 4, n4 = slot & 15;
      v4h h = pack4(br[i]);
      BsT[(n4 * 4 + 0) * BSTR + kk] = h[0];
      BsT[(n4 * 4 + 1) * BSTR + kk] = h[1];
      BsT[(n4 * 4 + 2) * BSTR + kk] = h[2];
      BsT[(n4 * 4 + 3) * BSTR + kk] = h[3];
    }
  };

  gload(0);
  sstore();
  for (int k0 = 0; k0 < K; k0 += 32) {
    __syncthreads();                       // current LDS tile ready
    if (k0 + 32 < K) gload(k0 + 32);       // prefetch next tile into registers

    const int arow = w * 32 + (L & 15);
    v16h a0 = frag_A(&As[arow * ASTR], L);
    v16h a1 = frag_A(&As[(arow + 16) * ASTR], L);
    for (int c = 0; c < 4; c++) {
      v16h bfr = frag_B(&BsT[(c * 16 + (L & 15)) * BSTR + lh16]);
      acc[0][c] = __builtin_amdgcn_wmma_f32_16x16x32_f16(
          false, a0, false, bfr, (short)0, acc[0][c], false, false);
      acc[1][c] = __builtin_amdgcn_wmma_f32_16x16x32_f16(
          false, a1, false, bfr, (short)0, acc[1][c], false, false);
    }
    __syncthreads();                       // compute done; safe to overwrite LDS
    if (k0 + 32 < K) sstore();
  }

  for (int m2 = 0; m2 < 2; m2++)
    for (int c = 0; c < 4; c++)
      for (int r = 0; r < 8; r++) {
        int row = m0 + w * 32 + m2 * 16 + r + ((L >> 4) << 3);
        int col = n0 + c * 16 + (L & 15);
        C[(size_t)row * N + col] = acc[m2][c][r] + bias[col];
      }
}

// =====================================================================================
// LIF: input currents I[s,b,h] = dot(Q[b,h,s,:], Wlif[h,:]) + blif[h]
// =====================================================================================
__global__ void lif_currents(const float* __restrict__ qkv,
                             const float* __restrict__ Wlif,
                             const float* __restrict__ blif,
                             float* __restrict__ I)
{
  int idx = blockIdx.x * blockDim.x + threadIdx.x;     // s*B*H + b*H + h
  if (idx >= Sn * Bn * Hn) return;
  int h = idx & (Hn - 1);
  int b = (idx / Hn) & (Bn - 1);
  int s = idx / (Bn * Hn);
  const float* q  = qkv + ((size_t)b * Sn + s) * (3 * En) + h * Dn;
  const float* wl = Wlif + h * Dn;
  float acc = blif[h];
  for (int d = 0; d < Dn; d += 4) {
    float4 qa = *(const float4*)&q[d];
    float4 wa = *(const float4*)&wl[d];
    acc += qa.x * wa.x + qa.y * wa.y + qa.z * wa.z + qa.w * wa.w;
  }
  I[idx] = acc;
}

// Sequential membrane scan (one wave, lane = b*16+h) + compaction of the
// "always visible" strided/global spiking columns per head.
__global__ void lif_scan(const float* __restrict__ I, float* __restrict__ maskH,
                         int* __restrict__ gidx, int* __restrict__ glen)
{
  int lane = threadIdx.x;          // 0..31
  int b = lane >> 4, h = lane & 15;
  const float decay = 1.f - 1.f / 10.f;   // TAU = 10
  float mem = 0.f;
  int cnt = 0;
  for (int s = 0; s < Sn; s++) {
    mem = mem * decay + I[(size_t)s * Bn * Hn + b * Hn + h];
    float spk = (mem >= 1.0f) ? 1.f : 0.f;              // THR = 1
    mem *= (1.f - spk);
    float other = __shfl_xor(spk, 16, 32);              // other batch, same head
    if (b == 0) {
      float comb = ((spk + other) > 0.f) ? 1.f : 0.f;
      maskH[h * Sn + s] = comb;
      if (comb > 0.f && ((s < NGLOBc) || ((s & (STRIDEc - 1)) == 0))) {
        gidx[h * GMAX + cnt] = s;
        cnt++;
      }
    }
  }
  if (b == 0) glen[h] = cnt;
}

// =====================================================================================
// Sparse flash attention. Block = (b, h, 64-query chunk), 4 waves x 16 query rows.
// Phase 1: <=2 gathered blocks of strided/global spiking columns (valid for all rows).
// Phase 2: 6 windowed key blocks covering [q0-WIN, q0+63], excluding phase-1 columns.
// =====================================================================================
__global__ __launch_bounds__(128) void spiking_attn(
    const float* __restrict__ qkv, const float* __restrict__ maskH,
    const int* __restrict__ gidx, const int* __restrict__ glen,
    float* __restrict__ attOut)
{
  constexpr int QS = 72;   // 64 d + 8 pad (halves)
  constexpr int KS = 72;
  constexpr int VS = 40;   // transposed layout [d][key]
  constexpr int PS = 40;
  __shared__ __align__(32) _Float16 qs [64 * QS];
  __shared__ __align__(32) _Float16 ks [32 * KS];
  __shared__ __align__(32) _Float16 vsT[64 * VS];
  __shared__ __align__(32) _Float16 pb [4][16 * PS];

  const int t = threadIdx.x, L = t & 31, w = t >> 5;
  const int q0 = blockIdx.x * 64;
  const int h  = blockIdx.y;
  const int b  = blockIdx.z;
  const int qw = q0 + w * 16;
  const int lh16 = (L >> 4) << 4;

  const size_t rstr = 3 * En;
  const float* Qb = qkv + (size_t)b * Sn * rstr + h * Dn;
  const float* Kb = Qb + En;
  const float* Vb = Qb + 2 * En;
  const int glenh = glen[h];

  // Stage 64x64 Q tile once; A-fragments live in registers for the whole kernel.
  for (int i = 0; i < 8; i++) {
    int slot = t + i * 128;
    int rr = slot >> 4, d4 = slot & 15;
    float4 f = *(const float4*)&Qb[(size_t)(q0 + rr) * rstr + d4 * 4];
    *(v4h*)&qs[rr * QS + d4 * 4] = pack4(f);
  }
  __syncthreads();
  const int qrow = w * 16 + (L & 15);
  v16h aQ0 = frag_A(&qs[qrow * QS], L);
  v16h aQ1 = frag_A(&qs[qrow * QS + 32], L);

  float mrow[8], lrow[8];
  v8f acc[4];
  for (int r = 0; r < 8; r++) { mrow[r] = -1e30f; lrow[r] = 0.f; }
  for (int c = 0; c < 4; c++)
    for (int r = 0; r < 8; r++) acc[c][r] = 0.f;

  const float scale = 0.125f;   // 1/sqrt(64)

  // Process one 32-key block. gathered=true: columns gidx[gbase..], all-row-valid.
  // gathered=false: contiguous keys [j0, j0+32) with window+spike mask, excluding
  // strided/global columns (those are covered by the gathered phase).
  auto process = [&](int j0, int gbase, bool gathered) {
    __syncthreads();
    for (int i = 0; i < 4; i++) {
      int slot = t + i * 128;
      int kk = slot >> 4, d4 = slot & 15;
      int row;
      if (gathered) {
        int gi = gbase + kk;
        row = (gi < glenh) ? gidx[h * GMAX + gi] : 0;
      } else {
        row = j0 + kk;
      }
      float4 fk = *(const float4*)&Kb[(size_t)row * rstr + d4 * 4];
      *(v4h*)&ks[kk * KS + d4 * 4] = pack4(fk);
      float4 fv = *(const float4*)&Vb[(size_t)row * rstr + d4 * 4];
      v4h hv = pack4(fv);
      vsT[(d4 * 4 + 0) * VS + kk] = hv[0];
      vsT[(d4 * 4 + 1) * VS + kk] = hv[1];
      vsT[(d4 * 4 + 2) * VS + kk] = hv[2];
      vsT[(d4 * 4 + 3) * VS + kk] = hv[3];
    }
    __syncthreads();

    float sc[2][8], tmax[8];
    for (int r = 0; r < 8; r++) tmax[r] = -1e30f;

    for (int st = 0; st < 2; st++) {
      const int key = st * 16 + (L & 15);
      v16h bK0 = frag_B(&ks[key * KS + lh16]);
      v16h bK1 = frag_B(&ks[key * KS + 32 + lh16]);
      v8f sA;
      for (int r = 0; r < 8; r++) sA[r] = 0.f;
      sA = __builtin_amdgcn_wmma_f32_16x16x32_f16(false, aQ0, false, bK0, (short)0, sA, false, false);
      sA = __builtin_amdgcn_wmma_f32_16x16x32_f16(false, aQ1, false, bK1, (short)0, sA, false, false);

      if (gathered) {
        bool ok = (gbase + key) < glenh;      // uniform over query rows
        for (int r = 0; r < 8; r++) {
          float s = ok ? sA[r] * scale : -1e30f;
          sc[st][r] = s;
          tmax[r] = fmaxf(tmax[r], s);
        }
      } else {
        int j = j0 + key;
        float sp = maskH[h * Sn + j];
        bool strided = ((j & (STRIDEc - 1)) == 0) || (j < NGLOBc);
        for (int r = 0; r < 8; r++) {
          int qi = qw + r + ((L >> 4) << 3);
          bool wind = (j <= qi) && (j >= qi - WINc);
          bool ok = (sp > 0.f) && wind && !strided;
          float s = ok ? sA[r] * scale : -1e30f;
          sc[st][r] = s;
          tmax[r] = fmaxf(tmax[r], s);
        }
      }
    }
    for (int off = 1; off < 16; off <<= 1)
      for (int r = 0; r < 8; r++) tmax[r] = fmaxf(tmax[r], __shfl_xor(tmax[r], off, 16));

    float rsum[8];
    for (int r = 0; r < 8; r++) {
      float nm = fmaxf(mrow[r], tmax[r]);
      float alpha = __expf(mrow[r] - nm);
      mrow[r] = nm;
      float e0 = __expf(sc[0][r] - nm);
      float e1 = __expf(sc[1][r] - nm);
      rsum[r] = e0 + e1;
      lrow[r] *= alpha;
      for (int c = 0; c < 4; c++) acc[c][r] *= alpha;
      int rowp = r + ((L >> 4) << 3);
      pb[w][rowp * PS + (L & 15)]      = (_Float16)e0;
      pb[w][rowp * PS + 16 + (L & 15)] = (_Float16)e1;
    }
    for (int off = 1; off < 16; off <<= 1)
      for (int r = 0; r < 8; r++) rsum[r] += __shfl_xor(rsum[r], off, 16);
    for (int r = 0; r < 8; r++) lrow[r] += rsum[r];

    __syncthreads();   // C-layout -> A-layout transpose of P through LDS

    v16h pA = frag_A(&pb[w][(L & 15) * PS], L);
    for (int c = 0; c < 4; c++) {
      v16h bV = frag_B(&vsT[(c * 16 + (L & 15)) * VS + lh16]);
      acc[c] = __builtin_amdgcn_wmma_f32_16x16x32_f16(
          false, pA, false, bV, (short)0, acc[c], false, false);
    }
  };

  // Phase 1: gathered strided/global spiking columns (<= 2 blocks, uniform per head)
  const int ng = (glenh + 31) / 32;
  for (int g = 0; g < ng; g++) process(0, g * 32, true);

  // Phase 2: windowed key blocks covering [q0-WIN, q0+63]
  const int lo = (q0 >= WINc) ? (q0 - WINc) : 0;
  for (int j0 = lo; j0 <= q0 + 32; j0 += 32) {
    if (j0 + 32 <= q0 + 32)   // prefetch next windowed K block
      if (t < 32) __builtin_prefetch(&Kb[(size_t)(j0 + 32 + t) * rstr], 0, 1);
    process(j0, 0, false);
  }

  // Epilogue: rows with no attendable keys -> 0 (matches reference nan_to_num(.,0))
  for (int r = 0; r < 8; r++) {
    int qi = qw + r + ((L >> 4) << 3);
    float inv = (mrow[r] > -1e29f && lrow[r] > 0.f) ? (1.f / lrow[r]) : 0.f;
    for (int c = 0; c < 4; c++)
      attOut[((size_t)b * Sn + qi) * En + h * Dn + c * 16 + (L & 15)] = acc[c][r] * inv;
  }
}

// =====================================================================================
extern "C" void kernel_launch(void* const* d_in, const int* in_sizes, int n_in,
                              void* d_out, int out_size, void* d_ws, size_t ws_size,
                              hipStream_t stream) {
  (void)in_sizes; (void)n_in; (void)out_size; (void)ws_size;
  const float* x    = (const float*)d_in[0];
  const float* Wqkv = (const float*)d_in[1];
  const float* bqkv = (const float*)d_in[2];
  const float* Wlif = (const float*)d_in[3];
  const float* blif = (const float*)d_in[4];
  const float* Wout = (const float*)d_in[5];
  const float* bout = (const float*)d_in[6];
  float* out = (float*)d_out;

  char* ws = (char*)d_ws;
  float* qkv   = (float*)(ws);                                     // [B,S,3E] f32
  float* att   = (float*)(ws + (size_t)Bn * Sn * 3 * En * 4);      // [B,S,E]  f32
  float* Icur  = (float*)(ws + (size_t)Bn * Sn * 4 * En * 4);      // [S,B,H]  f32
  float* maskH = (float*)((char*)Icur + (size_t)Sn * Bn * Hn * 4); // [H,S]    f32
  int*   gidx  = (int*)((char*)maskH + (size_t)Hn * Sn * 4);       // [H,GMAX]
  int*   glen  = (int*)((char*)gidx + (size_t)Hn * GMAX * 4);      // [H]

  const int M = Bn * Sn;   // 4096

  gemm_bias_f16<<<dim3((3 * En) / 64, M / 128), 128, 0, stream>>>(
      x, Wqkv, bqkv, qkv, M, 3 * En, En);
  lif_currents<<<(Sn * Bn * Hn) / 256, 256, 0, stream>>>(qkv, Wlif, blif, Icur);
  lif_scan<<<1, 32, 0, stream>>>(Icur, maskH, gidx, glen);
  spiking_attn<<<dim3(Sn / 64, Hn, Bn), 128, 0, stream>>>(qkv, maskH, gidx, glen, att);
  gemm_bias_f16<<<dim3(En / 64, M / 128), 128, 0, stream>>>(
      att, Wout, bout, out, M, En, En);
}
